// JobActor_74242804679194
// MI455X (gfx1250) — matrix operand order repaired and checked by reading, not compiled, not run on hardware
//
#include <hip/hip_runtime.h>

#define NB  8
#define NN  4096
#define IND 2
#define HID 64
#define NJ  128

typedef _Float16 h16;
typedef unsigned int u32;
typedef __attribute__((ext_vector_type(16))) _Float16 v16h;
typedef __attribute__((ext_vector_type(8)))  _Float16 v8h;
typedef __attribute__((ext_vector_type(8)))  float    v8f;
typedef __attribute__((ext_vector_type(4)))  float    v4f;
typedef __attribute__((ext_vector_type(4)))  u32      v4u;
typedef __attribute__((ext_vector_type(8)))  int      v8i;
typedef __attribute__((ext_vector_type(4)))  int      v4i;

static __device__ __forceinline__ v8f vzero8() {
  v8f z;
#pragma unroll
  for (int i = 0; i < 8; ++i) z[i] = 0.0f;
  return z;
}

// ---------------------------------------------------------------------------
// TDM: DMA a [64 rows x 32 halves] tile of h1t (row stride 4096 halves) into
// LDS at lds_off.  Descriptor per CDNA5 ISA:
//   group0: count=1 | lds_addr | global_addr[56:0] | type=2
//   group1: data_size=2B, tensor_dim0=4096, tensor_dim1=64,
//           tile_dim0=32, tile_dim1=64, tensor_dim0_stride=4096
//   groups 2/3: zero (2-D tensor)
// ---------------------------------------------------------------------------
static __device__ __forceinline__ void tdm_load_b_tile(const h16* gptr, u32 lds_off) {
  unsigned long long ga = (unsigned long long)(const void*)gptr;
  v4u g0;
  g0[0] = 1u;                                            // count=1, no gather
  g0[1] = lds_off;                                       // LDS byte address
  g0[2] = (u32)ga;                                       // global addr [31:0]
  g0[3] = ((u32)(ga >> 32) & 0x01FFFFFFu) | (2u << 30);  // addr[56:32] | type=2
  v8i g1;
  g1[0] = (int)(1u << 16);                               // data_size = 2 bytes
  g1[1] = (int)((NN & 0xFFFFu) << 16);                   // tensor_dim0 lo16
  g1[2] = (int)((NN >> 16) | ((HID & 0xFFFFu) << 16));   // dim0 hi | tensor_dim1 lo
  g1[3] = (int)((HID >> 16) | (32u << 16));              // dim1 hi | tile_dim0=32
  g1[4] = (int)HID;                                      // tile_dim1=64, tile_dim2=0
  g1[5] = (int)NN;                                       // tensor_dim0_stride lo32
  g1[6] = 0;                                             // stride hi | dim1_stride lo
  g1[7] = 0;
  v4i z4 = {0, 0, 0, 0};
#if defined(__clang_major__) && (__clang_major__ >= 23)
  v8i z8 = {0, 0, 0, 0, 0, 0, 0, 0};
  __builtin_amdgcn_tensor_load_to_lds(g0, g1, z4, z4, z8, 0);
#else
  __builtin_amdgcn_tensor_load_to_lds(g0, g1, z4, z4, 0);
#endif
}

// ---------------------------------------------------------------------------
// K0: transpose GIN-layer-1 weights to [out][in] f16 so WMMA B-tiles are
// contiguous 32B per lane.
// ---------------------------------------------------------------------------
__global__ void prep_weights(const float* __restrict__ w11,
                             const float* __restrict__ w21,
                             h16* __restrict__ w11t, h16* __restrict__ w21t) {
  for (int idx = threadIdx.x; idx < HID * HID; idx += blockDim.x) {
    int i = idx >> 6, o = idx & 63;               // w[i][o]
    w11t[o * HID + i] = (h16)w11[idx];
    w21t[o * HID + i] = (h16)w21[idx];
  }
}

// ---------------------------------------------------------------------------
// K1: layer 0 = (adj @ feat[N,2]) then MLP0 (2->64 relu ->64 relu), fused.
// One wave per node row; output written TRANSPOSED f16: h1t[b][h][node]
// (exactly the WMMA B-matrix memory order for layer 1).
// ---------------------------------------------------------------------------
__global__ __launch_bounds__(256)
void gin0_kernel(const float* __restrict__ adj, const float* __restrict__ feat,
                 const float* __restrict__ w1, const float* __restrict__ b1,
                 const float* __restrict__ w2, const float* __restrict__ b2,
                 h16* __restrict__ h1t) {
  __shared__ float feat_s[NN * IND];     // 32 KB
  __shared__ float w1s[IND * HID];
  __shared__ float w2s[HID * HID];       // 16 KB
  __shared__ float b1s[HID], b2s[HID];
  __shared__ float hid_s[8][HID];

  int tid = threadIdx.x;
  int b = blockIdx.x >> 9;               // 512 blocks per batch (8 rows each)
  int rowBlk = blockIdx.x & 511;

  const float* fb = feat + (size_t)b * NN * IND;
  for (int i = tid; i < NN * IND; i += 256) feat_s[i] = fb[i];
  for (int i = tid; i < IND * HID; i += 256) w1s[i] = w1[i];
  for (int i = tid; i < HID * HID; i += 256) w2s[i] = w2[i];
  if (tid < HID) { b1s[tid] = b1[tid]; b2s[tid] = b2[tid]; }
  __syncthreads();

  int wave = tid >> 5, lane = tid & 31;
  int row = rowBlk * 8 + wave;
  const float* ar = adj + ((size_t)b * NN + row) * (size_t)NN;

  float a0 = 0.f, a1 = 0.f;
  for (int k = lane * 4; k < NN; k += 128) {
    v4f v = *(const v4f*)(ar + k);
#pragma unroll
    for (int j = 0; j < 4; ++j) {
      a0 += v[j] * feat_s[(k + j) * 2];
      a1 += v[j] * feat_s[(k + j) * 2 + 1];
    }
  }
#pragma unroll
  for (int off = 16; off > 0; off >>= 1) {
    a0 += __shfl_xor(a0, off, 32);
    a1 += __shfl_xor(a1, off, 32);
  }
  // MLP0 layer 1 (2 -> 64)
  for (int o = lane; o < HID; o += 32)
    hid_s[wave][o] = fmaxf(a0 * w1s[o] + a1 * w1s[HID + o] + b1s[o], 0.f);
  // MLP0 layer 2 (64 -> 64), write transposed f16
  for (int o = lane; o < HID; o += 32) {
    float acc = b2s[o];
#pragma unroll 8
    for (int h = 0; h < HID; ++h) acc += hid_s[wave][h] * w2s[h * HID + o];
    h1t[((size_t)b * HID + o) * NN + row] = (h16)fmaxf(acc, 0.f);
  }
}

// ---------------------------------------------------------------------------
// K2: layer 1 = (adj @ h1) then MLP1 (64->64 relu ->64 relu), all via WMMA.
// One wave owns a 16-row x 64-col output tile.  adj is converted to f16 on
// the fly (values are exactly {0,1,2}).  The shared B-panel (h1t columns for
// the current K-window) is staged once per block into LDS by the Tensor Data
// Mover, double-buffered, synchronized with s_wait_tensorcnt + barrier.
// Fused MLP1 uses two more WMMA GEMMs with a C->A layout round trip through
// per-wave LDS scratch.
// ---------------------------------------------------------------------------
__global__ __launch_bounds__(256)
void gin1_wmma(const float* __restrict__ adj, const h16* __restrict__ h1t,
               const h16* __restrict__ w11t, const h16* __restrict__ w21t,
               const float* __restrict__ b11, const float* __restrict__ b21,
               float* __restrict__ h2) {
  __shared__ __attribute__((aligned(32))) h16 w1s[HID * HID];    // 8 KB
  __shared__ __attribute__((aligned(32))) h16 w2s[HID * HID];    // 8 KB
  __shared__ float b1s[HID], b2s[HID];
  __shared__ __attribute__((aligned(32))) h16 scr[8 * 16 * HID]; // 16 KB
  __shared__ __attribute__((aligned(32))) h16 bbuf[2][HID * 32]; // 2 x 4 KB

  int tid = threadIdx.x;
  for (int i = tid; i < HID * HID; i += 256) { w1s[i] = w11t[i]; w2s[i] = w21t[i]; }
  if (tid < HID) { b1s[tid] = b11[tid]; b2s[tid] = b21[tid]; }

  int wave = tid >> 5, lane = tid & 31;
  int tileG = blockIdx.x * 8 + wave;     // 2048 tiles total, 256 per batch
  int b = tileG >> 8;
  int tile = tileG & 255;
  int row0 = tile << 4;

  int lr = lane & 15;                    // local row (A) / column (B,C)
  int hi = lane >> 4;                    // half-wave select

  const float* arow = adj + ((size_t)b * NN + (row0 + lr)) * (size_t)NN;
  const h16* hb = h1t + (size_t)b * HID * NN;   // batch-uniform across block

  u32 lds_b0 = (u32)(size_t)(void*)&bbuf[0][0];
  u32 lds_b1 = (u32)(size_t)(void*)&bbuf[1][0];

  v8f acc[4];
#pragma unroll
  for (int ct = 0; ct < 4; ++ct) acc[ct] = vzero8();

  // Prologue: DMA the first B-panel, wait, release the block.
  if (wave == 0) {
    tdm_load_b_tile(hb, lds_b0);
    __builtin_amdgcn_s_wait_tensorcnt(0);
  }
  __syncthreads();

  // ---- SpMM: pooled1 = adj[16 x 4096] @ h1[4096 x 64] ----
  for (int kk = 0; kk < NN; kk += 32) {
    int cur = (kk >> 5) & 1;
    // kick off DMA of the next panel into the other buffer
    if (wave == 0 && (kk + 32) < NN)
      tdm_load_b_tile(hb + kk + 32, cur ? lds_b0 : lds_b1);

    // A tile: stream + convert adj
    int kbase = kk + (hi << 4);          // lanes 16-31 offset by 16 in K
    v4f f0 = *(const v4f*)(arow + kbase);
    v4f f1 = *(const v4f*)(arow + kbase + 4);
    v4f f2 = *(const v4f*)(arow + kbase + 8);
    v4f f3 = *(const v4f*)(arow + kbase + 12);
    __builtin_prefetch(arow + kk + 128, 0, 0);   // global_prefetch_b8
    v16h a;
#pragma unroll
    for (int i = 0; i < 4; ++i) {
      a[i]      = (h16)f0[i];
      a[4 + i]  = (h16)f1[i];
      a[8 + i]  = (h16)f2[i];
      a[12 + i] = (h16)f3[i];
    }
    // B tiles from the LDS panel (TDM wrote rows contiguously: [col][32])
    const h16* bp0 = &bbuf[cur][0];
    int kb = hi << 4;                    // lanes 16-31 take K +16..31
#pragma unroll
    for (int ct = 0; ct < 4; ++ct) {
      const v16h* bp = (const v16h*)(bp0 + (ct * 16 + lr) * 32 + kb);
      acc[ct] = __builtin_amdgcn_wmma_f32_16x16x32_f16(
          false, a, false, *bp, (short)0, acc[ct], false, false);
    }
    // next panel must be resident before anyone flips buffers
    if (wave == 0) __builtin_amdgcn_s_wait_tensorcnt(0);
    __syncthreads();
  }

  // ---- spill pooled1 C-tile to per-wave LDS scratch (row-major f16) ----
  h16* sw = scr + wave * 16 * HID;
#pragma unroll
  for (int ct = 0; ct < 4; ++ct) {
    int col = ct * 16 + lr;
#pragma unroll
    for (int r = 0; r < 8; ++r) sw[(hi * 8 + r) * HID + col] = (h16)acc[ct][r];
  }

  // ---- hidden = relu(pooled1 @ W1 + b1) via WMMA ----
  v8f hcc[4];
#pragma unroll
  for (int ct = 0; ct < 4; ++ct) hcc[ct] = vzero8();
#pragma unroll
  for (int kc = 0; kc < HID; kc += 32) {
    v16h a2;
    int kb2 = kc + (hi << 4);
    v8h lo = *(const v8h*)(sw + lr * HID + kb2);
    v8h hv = *(const v8h*)(sw + lr * HID + kb2 + 8);
#pragma unroll
    for (int i = 0; i < 8; ++i) { a2[i] = lo[i]; a2[8 + i] = hv[i]; }
    int kb3 = kc + (hi << 4);
#pragma unroll
    for (int ct = 0; ct < 4; ++ct) {
      const v16h* wp = (const v16h*)(w1s + (ct * 16 + lr) * HID + kb3);
      hcc[ct] = __builtin_amdgcn_wmma_f32_16x16x32_f16(
          false, a2, false, *wp, (short)0, hcc[ct], false, false);
    }
  }
#pragma unroll
  for (int ct = 0; ct < 4; ++ct) {
    int col = ct * 16 + lr;
    float bb = b1s[col];
#pragma unroll
    for (int r = 0; r < 8; ++r)
      sw[(hi * 8 + r) * HID + col] = (h16)fmaxf(hcc[ct][r] + bb, 0.f);
  }

  // ---- h2 = relu(hidden @ W2 + b2) via WMMA ----
  v8f occ[4];
#pragma unroll
  for (int ct = 0; ct < 4; ++ct) occ[ct] = vzero8();
#pragma unroll
  for (int kc = 0; kc < HID; kc += 32) {
    v16h a3;
    int kb2 = kc + (hi << 4);
    v8h lo = *(const v8h*)(sw + lr * HID + kb2);
    v8h hv = *(const v8h*)(sw + lr * HID + kb2 + 8);
#pragma unroll
    for (int i = 0; i < 8; ++i) { a3[i] = lo[i]; a3[8 + i] = hv[i]; }
    int kb3 = kc + (hi << 4);
#pragma unroll
    for (int ct = 0; ct < 4; ++ct) {
      const v16h* wp = (const v16h*)(w2s + (ct * 16 + lr) * HID + kb3);
      occ[ct] = __builtin_amdgcn_wmma_f32_16x16x32_f16(
          false, a3, false, *wp, (short)0, occ[ct], false, false);
    }
  }
#pragma unroll
  for (int ct = 0; ct < 4; ++ct) {
    int col = ct * 16 + lr;
    float bb = b2s[col];
#pragma unroll
    for (int r = 0; r < 8; ++r) {
      int m = hi * 8 + r;
      h2[((size_t)b * NN + row0 + m) * HID + col] = fmaxf(occ[ct][r] + bb, 0.f);
    }
  }
}

// ---------------------------------------------------------------------------
// K3: global graph embedding  gge[b][h] = sum_n gp[b][n] * h2[b][n][h]
// ---------------------------------------------------------------------------
__global__ __launch_bounds__(256)
void pool_kernel(const float* __restrict__ gp, const float* __restrict__ h2,
                 float* __restrict__ gge) {
  __shared__ float red[4][HID];
  int b = blockIdx.x, tid = threadIdx.x;
  int h = tid & 63, s = tid >> 6;
  float acc = 0.f;
  for (int n = s; n < NN; n += 4)
    acc += gp[(size_t)b * NN + n] * h2[((size_t)b * NN + n) * HID + h];
  red[s][h] = acc;
  __syncthreads();
  if (tid < HID)
    gge[b * HID + tid] = red[0][tid] + red[1][tid] + red[2][tid] + red[3][tid];
}

// ---------------------------------------------------------------------------
// K4: gather candidates, 3-layer tanh actor MLP, masked softmax over jobs.
// One block per batch, one wave per job (8 jobs per wave sequentially).
// ---------------------------------------------------------------------------
__global__ __launch_bounds__(512)
void actor_kernel(const float* __restrict__ h2, const float* __restrict__ gge,
                  const float* __restrict__ pmi, const int* __restrict__ cand,
                  const int* __restrict__ mask,
                  const float* __restrict__ aw1, const float* __restrict__ ab1,
                  const float* __restrict__ aw2, const float* __restrict__ ab2,
                  const float* __restrict__ aw3, const float* __restrict__ ab3,
                  float* __restrict__ out) {
  __shared__ float cs[16][3 * HID];
  __shared__ float a1s[16][HID];
  __shared__ float a2s[16][HID];
  __shared__ float w2l[HID * HID];
  __shared__ float b1l[HID], b2l[HID], w3l[HID];
  __shared__ float sc[NJ];
  __shared__ float mx_s, sum_s;

  int b = blockIdx.x, tid = threadIdx.x;
  int wave = tid >> 5, lane = tid & 31;
  for (int i = tid; i < HID * HID; i += 512) w2l[i] = aw2[i];
  if (tid < HID) { b1l[tid] = ab1[tid]; b2l[tid] = ab2[tid]; w3l[tid] = aw3[tid]; }
  __syncthreads();
  float b3v = ab3[0];

  for (int jj = 0; jj < NJ / 16; ++jj) {
    int j = wave + jj * 16;
    int node = cand[b * NJ + j];
    const float* emb = h2 + ((size_t)b * NN + node) * HID;
    for (int t = lane; t < HID; t += 32) {
      cs[wave][t]           = emb[t];
      cs[wave][HID + t]     = gge[b * HID + t];
      cs[wave][2 * HID + t] = pmi[t];
    }
    for (int o = lane; o < HID; o += 32) {
      float acc = b1l[o];
      for (int i = 0; i < 3 * HID; ++i) acc += cs[wave][i] * aw1[i * HID + o];
      a1s[wave][o] = tanhf(acc);
    }
    for (int o = lane; o < HID; o += 32) {
      float acc = b2l[o];
#pragma unroll 8
      for (int i = 0; i < HID; ++i) acc += a1s[wave][i] * w2l[i * HID + o];
      a2s[wave][o] = tanhf(acc);
    }
    float p = 0.f;
    for (int t = lane; t < HID; t += 32) p += a2s[wave][t] * w3l[t];
#pragma unroll
    for (int off = 16; off > 0; off >>= 1) p += __shfl_xor(p, off, 32);
    if (lane == 0) {
      float s = (p + b3v) * 10.f;
      if (mask[b * NJ + j]) s = -__builtin_inff();
      sc[j] = s;
    }
  }
  __syncthreads();
  if (tid == 0) {
    float m = -__builtin_inff();
    for (int j = 0; j < NJ; ++j) m = fmaxf(m, sc[j]);
    mx_s = m;
  }
  __syncthreads();
  if (tid < NJ) sc[tid] = expf(sc[tid] - mx_s);
  __syncthreads();
  if (tid == 0) {
    float s = 0.f;
    for (int j = 0; j < NJ; ++j) s += sc[j];
    sum_s = s;
  }
  __syncthreads();
  if (tid < NJ) out[b * NJ + tid] = sc[tid] / sum_s;
}

// ---------------------------------------------------------------------------
extern "C" void kernel_launch(void* const* d_in, const int* in_sizes, int n_in,
                              void* d_out, int out_size, void* d_ws, size_t ws_size,
                              hipStream_t stream) {
  const float* feat = (const float*)d_in[0];
  const float* gp   = (const float*)d_in[1];
  const float* adj  = (const float*)d_in[2];
  const int*   cand = (const int*)d_in[3];
  const int*   mask = (const int*)d_in[4];
  const float* g0w1 = (const float*)d_in[5];
  const float* g0b1 = (const float*)d_in[6];
  const float* g0w2 = (const float*)d_in[7];
  const float* g0b2 = (const float*)d_in[8];
  const float* g1w1 = (const float*)d_in[9];
  const float* g1b1 = (const float*)d_in[10];
  const float* g1w2 = (const float*)d_in[11];
  const float* g1b2 = (const float*)d_in[12];
  const float* pmi  = (const float*)d_in[13];
  const float* aw1  = (const float*)d_in[14];
  const float* ab1  = (const float*)d_in[15];
  const float* aw2  = (const float*)d_in[16];
  const float* ab2  = (const float*)d_in[17];
  const float* aw3  = (const float*)d_in[18];
  const float* ab3  = (const float*)d_in[19];

  char* ws = (char*)d_ws;
  h16*   h1t  = (h16*)(ws);                                  // 4 MB
  float* h2   = (float*)(ws + (4u << 20));                   // 8 MB
  float* gge  = (float*)(ws + (12u << 20));                  // 2 KB
  h16*   w11t = (h16*)(ws + (12u << 20) + 4096);             // 8 KB
  h16*   w21t = (h16*)(ws + (12u << 20) + 4096 + 8192);      // 8 KB

  prep_weights<<<1, 256, 0, stream>>>(g1w1, g1w2, w11t, w21t);
  gin0_kernel<<<NB * NN / 8, 256, 0, stream>>>(adj, feat, g0w1, g0b1, g0w2, g0b2, h1t);
  gin1_wmma<<<NB * (NN / 16) / 8, 256, 0, stream>>>(adj, h1t, w11t, w21t, g1b1, g1b2, h2);
  pool_kernel<<<NB, 256, 0, stream>>>(gp, h2, gge);
  actor_kernel<<<NB, 512, 0, stream>>>(h2, gge, pmi, cand, mask,
                                       aw1, ab1, aw2, ab2, aw3, ab3, (float*)d_out);
}
